// MultiHeadAttention_14688788152575
// MI455X (gfx1250) — matrix operand order repaired and checked by reading, not compiled, not run on hardware
//
#include <hip/hip_runtime.h>
#include <hip/hip_bf16.h>
#include <math.h>

#define DEVINL __device__ __forceinline__

typedef __attribute__((ext_vector_type(16))) _Float16 v16h;
typedef __attribute__((ext_vector_type(8)))  _Float16 v8h;
typedef __attribute__((ext_vector_type(8)))  float    v8f;

// Problem constants (from reference)
constexpr int CB  = 8;     // batch
constexpr int CS  = 1024;  // sequence
constexpr int CDM = 1024;  // d_model
constexpr int CH  = 8;     // heads
constexpr int CDK = 128;   // d_k = d_v

DEVINL v8f wmma_f16(v16h a, v16h b, v8f c) {
  // D = A(16x32 f16) * B(32x16 f16) + C(16x16 f32)
  return __builtin_amdgcn_wmma_f32_16x16x32_f16(
      /*neg_a=*/false, a, /*neg_b=*/false, b,
      /*c_mod=*/(short)0, c, /*reuse_a=*/false, /*reuse_b=*/false);
}

DEVINL v16h cvt16(float4 p0, float4 p1, float4 p2, float4 p3) {
  v16h r;
  r[0]=(_Float16)p0.x;  r[1]=(_Float16)p0.y;  r[2]=(_Float16)p0.z;  r[3]=(_Float16)p0.w;
  r[4]=(_Float16)p1.x;  r[5]=(_Float16)p1.y;  r[6]=(_Float16)p1.z;  r[7]=(_Float16)p1.w;
  r[8]=(_Float16)p2.x;  r[9]=(_Float16)p2.y;  r[10]=(_Float16)p2.z; r[11]=(_Float16)p2.w;
  r[12]=(_Float16)p3.x; r[13]=(_Float16)p3.y; r[14]=(_Float16)p3.z; r[15]=(_Float16)p3.w;
  return r;
}

DEVINL v16h cat16(v8h x, v8h y) {
  v16h r;
#pragma unroll
  for (int i = 0; i < 8; ++i) { r[i] = x[i]; r[i + 8] = y[i]; }
  return r;
}

// A fragment from f32 source: lane chunks K = {aoff..+7, aoff+16..+23}
DEVINL v16h loadA_f32(const float* p) {
  return cvt16(*(const float4*)(p),      *(const float4*)(p + 4),
               *(const float4*)(p + 16), *(const float4*)(p + 20));
}
// A fragment from f16 source
DEVINL v16h loadA_f16(const _Float16* p) {
  return cat16(*(const v8h*)(p), *(const v8h*)(p + 16));
}
// B fragment from f16 source: lane chunks K = {boff..+7, boff+16..+23}
DEVINL v16h loadB_f16(const _Float16* p) {
  return cat16(*(const v8h*)(p), *(const v8h*)(p + 16));
}

// ---------------------------------------------------------------------------
// One-shot f32 -> f16 conversion (8 elements per thread).
// ---------------------------------------------------------------------------
__global__ void cvt_f16_kernel(const float* __restrict__ src,
                               _Float16* __restrict__ dst, int n8) {
  const int i = blockIdx.x * blockDim.x + threadIdx.x;
  if (i >= n8) return;
  const float4 a = ((const float4*)src)[2 * i];
  const float4 b = ((const float4*)src)[2 * i + 1];
  v8h o;
  o[0]=(_Float16)a.x; o[1]=(_Float16)a.y; o[2]=(_Float16)a.z; o[3]=(_Float16)a.w;
  o[4]=(_Float16)b.x; o[5]=(_Float16)b.y; o[6]=(_Float16)b.z; o[7]=(_Float16)b.w;
  ((v8h*)dst)[i] = o;
}

// ---------------------------------------------------------------------------
// Projection: out[b,h,s,dk] = sum_m X[b,s,m] * W[h,dk,m] + bias[h,dk]
// f16 operands, 2x2 register blocking (32 s x 32 dk per wave).
// TRANS_OUT stores out^T as [b, dk, s] (f16) for the attn*V B-operand.
// ---------------------------------------------------------------------------
template <bool TRANS_OUT>
__global__ __launch_bounds__(32, 1)
void proj_kernel(const _Float16* __restrict__ X,
                 const _Float16* __restrict__ W,
                 const float* __restrict__ bias,
                 _Float16* __restrict__ out, int H) {
  const int lane = threadIdx.x;
  const int row  = lane & 15;
  const bool hi  = lane >= 16;
  const int s0   = blockIdx.x * 32;
  const int n0   = blockIdx.y * 32;
  const int bh   = blockIdx.z;
  const int b    = bh / H;
  const int h    = bh - b * H;

  const int aoff = hi ? 8 : 0;
  const int boff = hi ? 8 : 0;
  const _Float16* Xb = X + ((size_t)b * CS + s0 + row) * CDM + aoff;
  const _Float16* Wh = W + ((size_t)h * CDK + n0 + row) * CDM + boff;

  v8f acc[2][2] = {};
#pragma unroll 4
  for (int k0 = 0; k0 < CDM; k0 += 32) {
    v16h a[2], bm[2];
#pragma unroll
    for (int i = 0; i < 2; ++i) a[i]  = loadA_f16(Xb + (size_t)i * 16 * CDM + k0);
#pragma unroll
    for (int j = 0; j < 2; ++j) bm[j] = loadB_f16(Wh + (size_t)j * 16 * CDM + k0);
#pragma unroll
    for (int i = 0; i < 2; ++i)
#pragma unroll
      for (int j = 0; j < 2; ++j)
        acc[i][j] = wmma_f16(a[i], bm[j], acc[i][j]);
  }

  const int mb = hi ? 8 : 0;  // row offset of this half-wave's D rows
  const float bb0 = bias[(size_t)h * CDK + n0 + row];
  const float bb1 = bias[(size_t)h * CDK + n0 + 16 + row];

  if (TRANS_OUT) {
    // base + immediate offsets; 8 contiguous halves per (i,j) -> 16B stores
    _Float16* ob = out + ((size_t)b * CDK + n0 + row) * CS + s0 + mb;
#pragma unroll
    for (int j = 0; j < 2; ++j) {
      const float bb = j ? bb1 : bb0;
#pragma unroll
      for (int i = 0; i < 2; ++i) {
        v8h pk;
#pragma unroll
        for (int r = 0; r < 8; ++r) pk[r] = (_Float16)(acc[i][j][r] + bb);
        *(v8h*)(ob + (size_t)j * 16 * CS + i * 16) = pk;
      }
    }
  } else {
    _Float16* ob = out + (((size_t)bh) * CS + s0 + mb) * CDK + n0 + row;
#pragma unroll
    for (int i = 0; i < 2; ++i)
#pragma unroll
      for (int r = 0; r < 8; ++r) {
#pragma unroll
        for (int j = 0; j < 2; ++j)
          ob[(size_t)(i * 16 + r) * CDK + j * 16] =
              (_Float16)(acc[i][j][r] + (j ? bb1 : bb0));
      }
  }
}

// ---------------------------------------------------------------------------
// Scores: attn[b,q,h,k] = (Q . K) / sqrt(dk)
// 2x2 blocking (32 q x 32 keys per wave), written straight into d_out's attn
// region (layout [B,Sq,H,Sk]). Masking is deferred to the softmax kernel,
// where the mask row is contiguous (coalesced dword loads instead of 32
// scattered byte loads here).
// ---------------------------------------------------------------------------
__global__ __launch_bounds__(32, 1)
void scores_kernel(const _Float16* __restrict__ Q,
                   const _Float16* __restrict__ K,
                   float* __restrict__ attn) {
  const int lane = threadIdx.x;
  const int row  = lane & 15;
  const bool hi  = lane >= 16;
  const int q0   = blockIdx.x * 32;
  const int n0   = blockIdx.y * 32;
  const int bh   = blockIdx.z;
  const int b    = bh / CH;
  const int h    = bh - b * CH;

  const int aoff = hi ? 8 : 0;
  const int boff = hi ? 8 : 0;
  const _Float16* Qp = Q + ((size_t)bh * CS + q0 + row) * CDK + aoff;
  const _Float16* Kp = K + ((size_t)bh * CS + n0 + row) * CDK + boff;

  v8f acc[2][2] = {};
#pragma unroll
  for (int k0 = 0; k0 < CDK; k0 += 32) {
    v16h a[2], bm[2];
#pragma unroll
    for (int i = 0; i < 2; ++i) a[i]  = loadA_f16(Qp + (size_t)i * 16 * CDK + k0);
#pragma unroll
    for (int j = 0; j < 2; ++j) bm[j] = loadB_f16(Kp + (size_t)j * 16 * CDK + k0);
#pragma unroll
    for (int i = 0; i < 2; ++i)
#pragma unroll
      for (int j = 0; j < 2; ++j)
        acc[i][j] = wmma_f16(a[i], bm[j], acc[i][j]);
  }

  const int mb = hi ? 8 : 0;
  // Single base pointer; all 32 store offsets are compile-time immediates.
  float* obase = attn + (((size_t)(b * CS + q0 + mb)) * CH + h) * CS + n0 + row;
  const float scale = 0.08838834764831845f;  // 1/sqrt(128)
#pragma unroll
  for (int i = 0; i < 2; ++i)
#pragma unroll
    for (int r = 0; r < 8; ++r)
#pragma unroll
      for (int j = 0; j < 2; ++j)
        obase[(size_t)(i * 16 + r) * CH * CS + j * 16] = acc[i][j][r] * scale;
}

// ---------------------------------------------------------------------------
// Row softmax over the last dim (Sk = 1024), one wave32 per (b,q,h) row.
// Applies the boolean mask (-> -inf) before the max/exp, using coalesced
// dword loads of the contiguous mask row mask[b,q,:].
// ---------------------------------------------------------------------------
__global__ __launch_bounds__(32, 1)
void softmax_kernel(float* __restrict__ attn,
                    const unsigned char* __restrict__ mask) {
  const int rowid = blockIdx.x;          // over [B*Sq*H], attn row-major
  float* p = attn + (size_t)rowid * CS;
  const int bq = rowid / CH;             // b*CS + q
  const unsigned char* mrow = mask + (size_t)bq * CS;
  const int lane = threadIdx.x;

  float4 v[8];
  float mx = -INFINITY;
#pragma unroll
  for (int i = 0; i < 8; ++i) {
    v[i] = *(const float4*)(p + i * 128 + lane * 4);
    const uchar4 m = *(const uchar4*)(mrow + i * 128 + lane * 4);
    if (m.x) v[i].x = -INFINITY;
    if (m.y) v[i].y = -INFINITY;
    if (m.z) v[i].z = -INFINITY;
    if (m.w) v[i].w = -INFINITY;
    mx = fmaxf(mx, fmaxf(fmaxf(v[i].x, v[i].y), fmaxf(v[i].z, v[i].w)));
  }
#pragma unroll
  for (int d = 16; d >= 1; d >>= 1) mx = fmaxf(mx, __shfl_xor(mx, d, 32));

  float sum = 0.0f;
#pragma unroll
  for (int i = 0; i < 8; ++i) {
    v[i].x = __expf(v[i].x - mx); v[i].y = __expf(v[i].y - mx);
    v[i].z = __expf(v[i].z - mx); v[i].w = __expf(v[i].w - mx);
    sum += v[i].x + v[i].y + v[i].z + v[i].w;
  }
#pragma unroll
  for (int d = 16; d >= 1; d >>= 1) sum += __shfl_xor(sum, d, 32);

  const float inv = 1.0f / sum;
#pragma unroll
  for (int i = 0; i < 8; ++i) {
    v[i].x *= inv; v[i].y *= inv; v[i].z *= inv; v[i].w *= inv;
    *(float4*)(p + i * 128 + lane * 4) = v[i];
  }
}

// ---------------------------------------------------------------------------
// Hpart[b,h,q,dv] = sum_k attn[b,q,h,k] * Vt[b,dv,k]
// One wave per (b,h,q-tile): full dv=128 strip (8 N-tiles), so the 256MB
// attention tensor streams from HBM exactly once; each A fragment feeds
// 8 WMMAs. Vt (2MB) stays L2-resident.
// ---------------------------------------------------------------------------
__global__ __launch_bounds__(32, 1)
void attnv_kernel(const float* __restrict__ attn,
                  const _Float16* __restrict__ Vt,
                  float* __restrict__ Hpart) {
  const int lane = threadIdx.x;
  const int row  = lane & 15;
  const bool hi  = lane >= 16;
  const int q0   = blockIdx.x * 16;
  const int bh   = blockIdx.y;  // b*CH + h
  const int b    = bh / CH;
  const int h    = bh - b * CH;

  const int aoff = hi ? 8 : 0;
  const int boff = hi ? 8 : 0;
  const size_t arow = (size_t)CH * CS;  // stride between q rows in attn

  const float* Ap = attn + (((size_t)b * CS + q0) * CH + h) * CS +
                    (size_t)row * arow + aoff;
  const _Float16* Bp = Vt + ((size_t)b * CDK + row) * CS + boff;

  v8f acc[8] = {};
  for (int k0 = 0; k0 < CS; k0 += 32) {
    const v16h a = loadA_f32(Ap + k0);
#pragma unroll
    for (int nt = 0; nt < 8; ++nt) {
      const v16h bm = loadB_f16(Bp + (size_t)nt * 16 * CS + k0);
      acc[nt] = wmma_f16(a, bm, acc[nt]);
    }
  }

  const int mb = hi ? 8 : 0;
  float* ob = Hpart + ((size_t)bh * CS + q0 + mb) * CDK + row;
#pragma unroll
  for (int nt = 0; nt < 8; ++nt)
#pragma unroll
    for (int r = 0; r < 8; ++r)
      ob[(size_t)r * CDK + nt * 16] = acc[nt][r];
}

// ---------------------------------------------------------------------------
// O16[b,q,dv] = (1/H) * sum_h Hpart[b,h,q,dv]   (head mean, f16 output)
// ---------------------------------------------------------------------------
__global__ void headmean_kernel(const float* __restrict__ Hpart,
                                _Float16* __restrict__ O16) {
  const size_t i8 = (size_t)blockIdx.x * blockDim.x + threadIdx.x;  // 8-float id
  const size_t perB = (size_t)CS * CDK;  // floats per (b,h)
  const size_t b = (i8 * 8) / perB;
  const size_t rem = i8 * 8 - b * perB;

  const float* src = Hpart + b * CH * perB + rem;
  float s[8] = {0, 0, 0, 0, 0, 0, 0, 0};
#pragma unroll
  for (int h = 0; h < CH; ++h) {
    const float4 t0 = *(const float4*)(src + (size_t)h * perB);
    const float4 t1 = *(const float4*)(src + (size_t)h * perB + 4);
    s[0] += t0.x; s[1] += t0.y; s[2] += t0.z; s[3] += t0.w;
    s[4] += t1.x; s[5] += t1.y; s[6] += t1.z; s[7] += t1.w;
  }
  v8h o;
#pragma unroll
  for (int r = 0; r < 8; ++r) o[r] = (_Float16)(s[r] * 0.125f);
  ((v8h*)O16)[i8] = o;
}

// ---------------------------------------------------------------------------
// outputs[b,q,m] = sum_dv O16[b,q,dv] * Wo16[m,dv]   (f16 operands, 2x2)
// ---------------------------------------------------------------------------
__global__ __launch_bounds__(32, 1)
void outproj_kernel(const _Float16* __restrict__ O16,
                    const _Float16* __restrict__ Wo,
                    float* __restrict__ out) {
  const int lane = threadIdx.x;
  const int row  = lane & 15;
  const bool hi  = lane >= 16;
  const int q0   = blockIdx.x * 32;
  const int n0   = blockIdx.y * 32;
  const int b    = blockIdx.z;

  const int aoff = hi ? 8 : 0;
  const int boff = hi ? 8 : 0;
  const _Float16* Ap = O16 + ((size_t)b * CS + q0 + row) * CDK + aoff;
  const _Float16* Bp = Wo + (size_t)(n0 + row) * CDK + boff;

  v8f acc[2][2] = {};
#pragma unroll
  for (int k0 = 0; k0 < CDK; k0 += 32) {
    v16h a[2], bm[2];
#pragma unroll
    for (int i = 0; i < 2; ++i) a[i]  = loadA_f16(Ap + (size_t)i * 16 * CDK + k0);
#pragma unroll
    for (int j = 0; j < 2; ++j) bm[j] = loadB_f16(Bp + (size_t)j * 16 * CDK + k0);
#pragma unroll
    for (int i = 0; i < 2; ++i)
#pragma unroll
      for (int j = 0; j < 2; ++j)
        acc[i][j] = wmma_f16(a[i], bm[j], acc[i][j]);
  }

  const int mb = hi ? 8 : 0;
  float* ob = out + ((size_t)b * CS + q0 + mb) * CDM + n0 + row;
#pragma unroll
  for (int i = 0; i < 2; ++i)
#pragma unroll
    for (int r = 0; r < 8; ++r)
#pragma unroll
      for (int j = 0; j < 2; ++j)
        ob[(size_t)(i * 16 + r) * CDM + j * 16] = acc[i][j][r];
}

// ---------------------------------------------------------------------------
extern "C" void kernel_launch(void* const* d_in, const int* in_sizes, int n_in,
                              void* d_out, int out_size, void* d_ws, size_t ws_size,
                              hipStream_t stream) {
  const float* x_query = (const float*)d_in[0];
  const float* x_key   = (const float*)d_in[1];
  const float* x_value = (const float*)d_in[2];
  const unsigned char* mask = (const unsigned char*)d_in[3];  // jnp bool -> 1 byte
  const float* Wq = (const float*)d_in[4];
  const float* bq = (const float*)d_in[5];
  const float* Wk = (const float*)d_in[6];
  const float* bk = (const float*)d_in[7];
  const float* Wv = (const float*)d_in[8];
  const float* bv = (const float*)d_in[9];
  const float* Wo = (const float*)d_in[10];

  // Workspace layout (f16 elements unless noted; all regions fully written
  // before being read):
  const size_t nX  = (size_t)CB * CS * CDM;      // 8M   (x tensors)
  const size_t nWqk = (size_t)CH * CDK * CDM;    // 1M   (Wq/Wk)
  const size_t nWv = (size_t)CDK * CDM;          // 128K (Wv / Wo)
  const size_t nQK = (size_t)CB * CH * CS * CDK; // 8M   (Q/K/Hpart)
  const size_t nV  = (size_t)CB * CDK * CS;      // 1M   (Vt)
  const size_t nO  = (size_t)CB * CS * CDK;      // 1M   (O16)

  _Float16* p = (_Float16*)d_ws;
  _Float16* Xq16 = p; p += nX;
  _Float16* Xk16 = p; p += nX;
  _Float16* Xv16 = p; p += nX;
  _Float16* Wq16 = p; p += nWqk;
  _Float16* Wk16 = p; p += nWqk;
  _Float16* Wv16 = p; p += nWv;
  _Float16* Wo16 = p; p += nWv;
  _Float16* Qh   = p; p += nQK;
  _Float16* Kh   = p; p += nQK;
  _Float16* Vt   = p; p += nV;
  _Float16* O16  = p; p += nO;
  float* Hpart = (float*)p;  // nQK f32 (32MB)

  float* out_main = (float*)d_out;                    // [B, S, Dm]
  float* attn     = out_main + (size_t)CB * CS * CDM; // [B, Sq, H, Sk]

  dim3 blk(32);
  dim3 cblk(256);

  // f32 -> f16 one-shot conversions
  cvt_f16_kernel<<<dim3((int)(nX / 8 / 256)), cblk, 0, stream>>>(x_query, Xq16, (int)(nX / 8));
  cvt_f16_kernel<<<dim3((int)(nX / 8 / 256)), cblk, 0, stream>>>(x_key,   Xk16, (int)(nX / 8));
  cvt_f16_kernel<<<dim3((int)(nX / 8 / 256)), cblk, 0, stream>>>(x_value, Xv16, (int)(nX / 8));
  cvt_f16_kernel<<<dim3((int)(nWqk / 8 / 256)), cblk, 0, stream>>>(Wq, Wq16, (int)(nWqk / 8));
  cvt_f16_kernel<<<dim3((int)(nWqk / 8 / 256)), cblk, 0, stream>>>(Wk, Wk16, (int)(nWqk / 8));
  cvt_f16_kernel<<<dim3((int)(nWv / 8 / 256)), cblk, 0, stream>>>(Wv, Wv16, (int)(nWv / 8));
  cvt_f16_kernel<<<dim3((int)(nWv / 8 / 256)), cblk, 0, stream>>>(Wo, Wo16, (int)(nWv / 8));

  proj_kernel<false><<<dim3(CS / 32, CDK / 32, CB * CH), blk, 0, stream>>>(
      Xq16, Wq16, bq, Qh, CH);
  proj_kernel<false><<<dim3(CS / 32, CDK / 32, CB * CH), blk, 0, stream>>>(
      Xk16, Wk16, bk, Kh, CH);
  proj_kernel<true><<<dim3(CS / 32, CDK / 32, CB), blk, 0, stream>>>(
      Xv16, Wv16, bv, Vt, 1);

  scores_kernel<<<dim3(CS / 32, CS / 32, CB * CH), blk, 0, stream>>>(
      Qh, Kh, attn);

  softmax_kernel<<<dim3(CB * CS * CH), blk, 0, stream>>>(attn, mask);

  attnv_kernel<<<dim3(CS / 16, CB * CH), blk, 0, stream>>>(attn, Vt, Hpart);

  headmean_kernel<<<dim3((int)(nO / 8 / 256)), cblk, 0, stream>>>(Hpart, O16);

  outproj_kernel<<<dim3(CS / 32, CDM / 32, CB), blk, 0, stream>>>(
      O16, Wo16, out_main);
}
